// EncoderBlock_43361989820847
// MI455X (gfx1250) — matrix operand order repaired and checked by reading, not compiled, not run on hardware
//
#include <hip/hip_runtime.h>

typedef __bf16 bf16;
typedef __attribute__((ext_vector_type(8)))  bf16  bf16x8;
typedef __attribute__((ext_vector_type(16))) bf16  bf16x16;
typedef __attribute__((ext_vector_type(8)))  float f32x8;
typedef __attribute__((ext_vector_type(4)))  unsigned int u32x4;
typedef __attribute__((ext_vector_type(4)))  int   i32x4;
typedef __attribute__((ext_vector_type(8)))  int   i32x8;

__device__ __forceinline__ bf16x16 cat8(bf16x8 a, bf16x8 b) {
  return __builtin_shufflevector(a, b, 0,1,2,3,4,5,6,7,8,9,10,11,12,13,14,15);
}
__device__ __forceinline__ f32x8 wmma_bf16(bf16x16 a, bf16x16 b, f32x8 c) {
  return __builtin_amdgcn_wmma_f32_16x16x32_bf16(false, a, false, b, (short)0, c,
                                                 false, false);
}

// ---------------------------------------------------------------------------
// Tensor Data Mover support (guarded; fallback = synchronous loads)
// ---------------------------------------------------------------------------
#if __has_builtin(__builtin_amdgcn_tensor_load_to_lds)
#define USE_TDM 1
#if __has_include(<hip/amd_detail/amd_gfx1250_TDM.h>)
#define TDM_LOAD(g0, g1, g2, g3) \
  __builtin_amdgcn_tensor_load_to_lds((g0), (g1), (g2), (g3), \
                                      (i32x8){0, 0, 0, 0, 0, 0, 0, 0}, 0)
#else
#define TDM_LOAD(g0, g1, g2, g3) \
  __builtin_amdgcn_tensor_load_to_lds((g0), (g1), (g2), (g3), 0)
#endif
#else
#define USE_TDM 0
#endif

#if __has_builtin(__builtin_amdgcn_s_wait_tensorcnt)
#define WAIT_TENSOR(n) __builtin_amdgcn_s_wait_tensorcnt(n)
#else
#define WAIT_TENSOR(n) asm volatile("s_wait_tensorcnt %0" ::"i"(n) : "memory")
#endif

#if USE_TDM
// 2-D bf16 tile DMA: global (row-major, `stride` elements/row) -> LDS.
// LDS padding: after every 2^(pad_icode+1) DWORDs stored, skip (pad_acode+1) DWORDs.
__device__ __forceinline__ void tdm_load_2d_bf16(const void* gptr, void* ldsptr,
                                                 int tile_w, int tile_h, int stride,
                                                 int pad_icode, int pad_acode) {
  unsigned long long ga = (unsigned long long)(uintptr_t)gptr;
  u32x4 g0;
  g0[0] = 1u;                                            // count=1, no gather
  g0[1] = (unsigned)(uintptr_t)ldsptr;                   // LDS byte offset
  g0[2] = (unsigned)ga;                                  // global addr [31:0]
  g0[3] = (unsigned)((ga >> 32) & 0x1ffffffu) | (2u << 30);  // addr[56:32] | type=2
  i32x8 g1;
  g1[0] = (1 << 16) | (1 << 20) | (pad_icode << 22) | (pad_acode << 25);
  g1[1] = (tile_w & 0xffff) << 16;                        // tensor_dim0[15:0]
  g1[2] = ((tile_w >> 16) & 0xffff) | ((tile_h & 0xffff) << 16);
  g1[3] = ((tile_h >> 16) & 0xffff) | ((tile_w & 0xffff) << 16);
  g1[4] = (tile_h & 0xffff);                              // tile_dim1 (tile_dim2=0)
  g1[5] = stride;                                         // tensor_dim0_stride lo32
  g1[6] = 0;
  g1[7] = 0;
  i32x4 z4 = {0, 0, 0, 0};
  TDM_LOAD(g0, g1, z4, z4);
}
#endif

// ---------------------------------------------------------------------------
// Elementwise f32 -> bf16
// ---------------------------------------------------------------------------
__global__ void __launch_bounds__(256) cvt_bf16_kernel(const float* __restrict__ x,
                                                       bf16* __restrict__ y, int n) {
  int i = (blockIdx.x * 256 + threadIdx.x) * 4;
  if (i + 3 < n) {
    float4 v = *(const float4*)(x + i);
    y[i + 0] = (bf16)v.x;
    y[i + 1] = (bf16)v.y;
    y[i + 2] = (bf16)v.z;
    y[i + 3] = (bf16)v.w;
  }
}

// ---------------------------------------------------------------------------
// Transpose + convert: W[K,N] f32 -> Wt[N,K] bf16 (32x32 LDS tiles)
// ---------------------------------------------------------------------------
__global__ void __launch_bounds__(256) transpose_cvt_kernel(const float* __restrict__ W,
                                                            bf16* __restrict__ Wt,
                                                            int K, int N) {
  __shared__ float tile[32][33];
  int n0 = blockIdx.x * 32, k0 = blockIdx.y * 32;
  int tx = threadIdx.x, ty = threadIdx.y;  // 32 x 8
#pragma unroll
  for (int i = 0; i < 32; i += 8)
    tile[ty + i][tx] = W[(size_t)(k0 + ty + i) * N + n0 + tx];
  __syncthreads();
#pragma unroll
  for (int i = 0; i < 32; i += 8)
    Wt[(size_t)(n0 + ty + i) * K + k0 + tx] = (bf16)tile[tx][ty + i];
}

// ---------------------------------------------------------------------------
// 8-WMMA stage: each wave computes its 32x32 patch over a 64-deep k stage.
// ---------------------------------------------------------------------------
__device__ __forceinline__ void gemm_tile_compute(const bf16* Ab, const bf16* Bb,
                                                  int wm, int wn, int lr, int hi,
                                                  f32x8 (&acc)[2][2]) {
  bf16x16 af[2][2], bfm[2][2];
#pragma unroll
  for (int mi = 0; mi < 2; ++mi)
#pragma unroll
    for (int c = 0; c < 2; ++c) {
      const bf16* p = &Ab[(wm * 32 + mi * 16 + lr) * 72 + c * 32 + hi * 8];
      af[mi][c] = cat8(*(const bf16x8*)p, *(const bf16x8*)(p + 16));
    }
#pragma unroll
  for (int ni = 0; ni < 2; ++ni)
#pragma unroll
    for (int c = 0; c < 2; ++c) {
      const bf16* p = &Bb[(wn * 32 + ni * 16 + lr) * 72 + c * 32 + hi * 8];
      bfm[ni][c] = cat8(*(const bf16x8*)p, *(const bf16x8*)(p + 16));
    }
#pragma unroll
  for (int mi = 0; mi < 2; ++mi)
#pragma unroll
    for (int ni = 0; ni < 2; ++ni)
#pragma unroll
      for (int c = 0; c < 2; ++c)
        acc[mi][ni] = wmma_bf16(af[mi][c], bfm[ni][c], acc[mi][ni]);
}

// ---------------------------------------------------------------------------
// bf16 GEMM: C[M,N] = A[M,K] * Bt[N,K]^T + bias
// MODE 0: f32 | MODE 1: bf16 | MODE 2: relu->bf16 | MODE 3: bf16, *0.125 (Q)
// Block tile 128x64, 8 waves, K step 64.
// TDM path: double-buffered tensor_load_to_lds pipeline driven by wave 0.
// ---------------------------------------------------------------------------
template <int MODE>
__global__ void __launch_bounds__(256)
gemm_bf16_kernel(const bf16* __restrict__ A, const bf16* __restrict__ Bt,
                 const float* __restrict__ bias, float* __restrict__ outF,
                 bf16* __restrict__ outB, int M, int N, int K) {
  __shared__ __align__(16) bf16 As[2 * 128 * 72];  // 128 rows x 64 k (+8 pad)
  __shared__ __align__(16) bf16 Bs[2 * 64 * 72];   // 64 rows  x 64 k (+8 pad)

  const int tid = threadIdx.x;
  const int m0 = blockIdx.y * 128, n0 = blockIdx.x * 64;
  const int lane = tid & 31, lr = lane & 15, hi = lane >> 4;
  const int wv = tid >> 5, wm = wv & 3, wn = wv >> 2;

  f32x8 acc[2][2];
  const f32x8 z8 = {0.f, 0.f, 0.f, 0.f, 0.f, 0.f, 0.f, 0.f};
  acc[0][0] = z8; acc[0][1] = z8; acc[1][0] = z8; acc[1][1] = z8;

  const int nk = K / 64;

#if USE_TDM
  // 64-elem (128B = 32 DWORD) rows, pad 8 elems (4 DWORDs): icode 4, acode 3
  if (wv == 0) {
    tdm_load_2d_bf16(A + (size_t)m0 * K, &As[0], 64, 128, K, 4, 3);
    tdm_load_2d_bf16(Bt + (size_t)n0 * K, &Bs[0], 64, 64, K, 4, 3);
  }
  for (int kc = 0; kc < nk; ++kc) {
    const int cur = kc & 1;
    if (wv == 0) {
      if (kc + 1 < nk) {
        tdm_load_2d_bf16(A + (size_t)m0 * K + (kc + 1) * 64, &As[(cur ^ 1) * 128 * 72],
                         64, 128, K, 4, 3);
        tdm_load_2d_bf16(Bt + (size_t)n0 * K + (kc + 1) * 64, &Bs[(cur ^ 1) * 64 * 72],
                         64, 64, K, 4, 3);
        WAIT_TENSOR(2);  // in-order: tiles for kc are complete
      } else {
        WAIT_TENSOR(0);
      }
    }
    __syncthreads();
    gemm_tile_compute(&As[cur * 128 * 72], &Bs[cur * 64 * 72], wm, wn, lr, hi, acc);
    __syncthreads();
  }
#else
  const int arow = tid >> 1, aseg = tid & 1;  // 128 rows x 2 segs of 32
  const int brow = tid >> 2, bseg = tid & 3;  // 64 rows  x 4 segs of 16
  const bf16* aptr = A + (size_t)(m0 + arow) * K + aseg * 32;
  const bf16* bptr = Bt + (size_t)(n0 + brow) * K + bseg * 16;
  bf16x8 a[4], b[2];
#pragma unroll
  for (int i = 0; i < 4; ++i) a[i] = *(const bf16x8*)(aptr + i * 8);
#pragma unroll
  for (int i = 0; i < 2; ++i) b[i] = *(const bf16x8*)(bptr + i * 8);
  for (int kc = 0; kc < nk; ++kc) {
#pragma unroll
    for (int i = 0; i < 4; ++i) *(bf16x8*)&As[arow * 72 + aseg * 32 + i * 8] = a[i];
#pragma unroll
    for (int i = 0; i < 2; ++i) *(bf16x8*)&Bs[brow * 72 + bseg * 16 + i * 8] = b[i];
    __syncthreads();
    if (kc + 1 < nk) {  // prefetch next stage while WMMAs run
#pragma unroll
      for (int i = 0; i < 4; ++i) a[i] = *(const bf16x8*)(aptr + (kc + 1) * 64 + i * 8);
#pragma unroll
      for (int i = 0; i < 2; ++i) b[i] = *(const bf16x8*)(bptr + (kc + 1) * 64 + i * 8);
    }
    gemm_tile_compute(&As[0], &Bs[0], wm, wn, lr, hi, acc);
    __syncthreads();
  }
#endif

#pragma unroll
  for (int mi = 0; mi < 2; ++mi)
#pragma unroll
    for (int ni = 0; ni < 2; ++ni) {
      int col = n0 + wn * 32 + ni * 16 + lr;
      float bv = bias[col];
#pragma unroll
      for (int r = 0; r < 8; ++r) {
        int row = m0 + wm * 32 + mi * 16 + r + hi * 8;
        float v = acc[mi][ni][r] + bv;
        if (MODE == 2) v = fmaxf(v, 0.f);
        if (MODE == 3) v *= 0.125f;  // fold 1/sqrt(Dk) into Q projection
        if (MODE == 0)
          outF[(size_t)row * N + col] = v;
        else
          outB[(size_t)row * N + col] = (bf16)v;
      }
    }
}

// ---------------------------------------------------------------------------
// Flash attention: one block = 64 Q rows of one (b, head); 4 waves x 16 rows.
// K tile via TDM (wave 0) overlapped with manual V^T staging; online softmax.
// Q is pre-scaled by 1/sqrt(Dk) in its projection GEMM (MODE 3).
// ---------------------------------------------------------------------------
__global__ void __launch_bounds__(128)
flash_attn_kernel(const bf16* __restrict__ q, const bf16* __restrict__ k,
                  const bf16* __restrict__ v, const int* __restrict__ mask,
                  bf16* __restrict__ o, int S, int D, int H) {
  __shared__ __align__(16) bf16 Ks[64 * 72];       // [t][d]
  __shared__ __align__(16) bf16 Vt[64 * 72];       // [d][t]  (transposed)
  __shared__ __align__(16) bf16 Pls[4 * 16 * 72];  // per-wave P staging [m][t]

  const int bh = blockIdx.y, b = bh / H, hd = bh % H;
  const int q0 = blockIdx.x * 64;
  const int tid = threadIdx.x;
  const int w = tid >> 5, lane = tid & 31, lr = lane & 15, hi = lane >> 4;
  const int rowbase = b * S;

  // Q fragments (A layout), 16 rows x 64 d per wave
  bf16x16 qf[2];
#pragma unroll
  for (int c = 0; c < 2; ++c) {
    const bf16* p = q + (size_t)(rowbase + q0 + w * 16 + lr) * D + hd * 64 + c * 32 + hi * 8;
    qf[c] = cat8(*(const bf16x8*)p, *(const bf16x8*)(p + 16));
  }

  f32x8 oacc[4];
  const f32x8 z8 = {0.f, 0.f, 0.f, 0.f, 0.f, 0.f, 0.f, 0.f};
  oacc[0] = z8; oacc[1] = z8; oacc[2] = z8; oacc[3] = z8;
  float mrow[8], lrow[8];
#pragma unroll
  for (int r = 0; r < 8; ++r) { mrow[r] = -1e30f; lrow[r] = 0.f; }

  const int nT = S / 64;
  for (int j = 0; j < nT; ++j) {
    const int t0 = j * 64;
#if USE_TDM
    if (w == 0)
      tdm_load_2d_bf16(k + (size_t)(rowbase + t0) * D + hd * 64, &Ks[0], 64, 64, D, 4, 3);
#endif
    {  // stage V transposed (all threads); K too when no TDM
      int trow = tid >> 1, seg = tid & 1;
      const bf16* vp = v + (size_t)(rowbase + t0 + trow) * D + hd * 64 + seg * 32;
#if !USE_TDM
      const bf16* kp = k + (size_t)(rowbase + t0 + trow) * D + hd * 64 + seg * 32;
#endif
#pragma unroll
      for (int i = 0; i < 4; ++i) {
#if !USE_TDM
        bf16x8 kv = *(const bf16x8*)(kp + i * 8);
        *(bf16x8*)&Ks[trow * 72 + seg * 32 + i * 8] = kv;
#endif
        bf16x8 vv = *(const bf16x8*)(vp + i * 8);
#pragma unroll
        for (int e = 0; e < 8; ++e)
          Vt[(seg * 32 + i * 8 + e) * 72 + trow] = vv[e];
      }
      if (j + 1 < nT) {  // prefetch hint for next tile
        __builtin_prefetch(vp + (size_t)64 * D, 0, 0);
        __builtin_prefetch(k + (size_t)(rowbase + t0 + 64 + trow) * D + hd * 64 + seg * 32, 0, 0);
      }
    }
#if USE_TDM
    if (w == 0) WAIT_TENSOR(0);
#endif
    __syncthreads();

    // hoist mask for this key tile
    int mk[4];
#pragma unroll
    for (int ni = 0; ni < 4; ++ni) mk[ni] = mask[b * S + t0 + ni * 16 + lr];

    // preload all K fragments, then back-to-back WMMAs: S = Q K^T
    bf16x16 kf[4][2];
#pragma unroll
    for (int ni = 0; ni < 4; ++ni)
#pragma unroll
      for (int c = 0; c < 2; ++c) {
        const bf16* p = &Ks[(ni * 16 + lr) * 72 + c * 32 + hi * 8];
        kf[ni][c] = cat8(*(const bf16x8*)p, *(const bf16x8*)(p + 16));
      }
    f32x8 s[4];
#pragma unroll
    for (int ni = 0; ni < 4; ++ni) {
      s[ni] = wmma_bf16(qf[0], kf[ni][0], z8);
      s[ni] = wmma_bf16(qf[1], kf[ni][1], s[ni]);
    }
#pragma unroll
    for (int ni = 0; ni < 4; ++ni)
#pragma unroll
      for (int r = 0; r < 8; ++r)
        s[ni][r] = mk[ni] ? s[ni][r] : -3.0e38f;

    // online softmax (row stats across 16 lanes of each half)
#pragma unroll
    for (int r = 0; r < 8; ++r) {
      float rm = fmaxf(fmaxf(s[0][r], s[1][r]), fmaxf(s[2][r], s[3][r]));
      rm = fmaxf(rm, __shfl_xor(rm, 1));
      rm = fmaxf(rm, __shfl_xor(rm, 2));
      rm = fmaxf(rm, __shfl_xor(rm, 4));
      rm = fmaxf(rm, __shfl_xor(rm, 8));
      float mnew = fmaxf(mrow[r], rm);
      float alpha = __expf(mrow[r] - mnew);
      float rs = 0.f;
#pragma unroll
      for (int ni = 0; ni < 4; ++ni) {
        float p = __expf(s[ni][r] - mnew);
        s[ni][r] = p;
        rs += p;
      }
      rs += __shfl_xor(rs, 1);
      rs += __shfl_xor(rs, 2);
      rs += __shfl_xor(rs, 4);
      rs += __shfl_xor(rs, 8);
      lrow[r] = lrow[r] * alpha + rs;
      mrow[r] = mnew;
#pragma unroll
      for (int ni = 0; ni < 4; ++ni) oacc[ni][r] *= alpha;
    }

    // P: C-layout -> A-layout via wave-private LDS
    bf16* pw = &Pls[w * 16 * 72];
#pragma unroll
    for (int ni = 0; ni < 4; ++ni)
#pragma unroll
      for (int r = 0; r < 8; ++r)
        pw[(r + hi * 8) * 72 + ni * 16 + lr] = (bf16)s[ni][r];
    asm volatile("s_wait_dscnt 0" ::: "memory");

    // preload P and V fragments, then back-to-back WMMAs: O += P V
    bf16x16 pf[2], vf[4][2];
#pragma unroll
    for (int c = 0; c < 2; ++c) {
      const bf16* pp = &pw[lr * 72 + c * 32 + hi * 8];
      pf[c] = cat8(*(const bf16x8*)pp, *(const bf16x8*)(pp + 16));
    }
#pragma unroll
    for (int ni = 0; ni < 4; ++ni)
#pragma unroll
      for (int c = 0; c < 2; ++c) {
        const bf16* vp2 = &Vt[(ni * 16 + lr) * 72 + c * 32 + hi * 8];
        vf[ni][c] = cat8(*(const bf16x8*)vp2, *(const bf16x8*)(vp2 + 16));
      }
#pragma unroll
    for (int ni = 0; ni < 4; ++ni) {
      oacc[ni] = wmma_bf16(pf[0], vf[ni][0], oacc[ni]);
      oacc[ni] = wmma_bf16(pf[1], vf[ni][1], oacc[ni]);
    }
    __syncthreads();
  }

#pragma unroll
  for (int ni = 0; ni < 4; ++ni)
#pragma unroll
    for (int r = 0; r < 8; ++r) {
      int row = rowbase + q0 + w * 16 + r + hi * 8;
      int col = hd * 64 + ni * 16 + lr;
      o[(size_t)row * D + col] = (bf16)(oacc[ni][r] / lrow[r]);
    }
}

// ---------------------------------------------------------------------------
// Residual add + LayerNorm, block per row (D = 1024 = 256 threads x 4)
// ---------------------------------------------------------------------------
__global__ void __launch_bounds__(256)
ln_residual_kernel(const float* __restrict__ x, const float* __restrict__ r,
                   const float* __restrict__ g, const float* __restrict__ be,
                   float* __restrict__ outF, bf16* __restrict__ outB, int D) {
  const int row = blockIdx.x, t = threadIdx.x;
  const float* xr = x + (size_t)row * D;
  const float* rr = r + (size_t)row * D;
  float v[4];
  float s = 0.f;
#pragma unroll
  for (int i = 0; i < 4; ++i) {
    int c = t + i * 256;
    v[i] = xr[c] + rr[c];
    s += v[i];
  }
  __shared__ float red[8];
  s += __shfl_xor(s, 1); s += __shfl_xor(s, 2); s += __shfl_xor(s, 4);
  s += __shfl_xor(s, 8); s += __shfl_xor(s, 16);
  int wv = t >> 5;
  if ((t & 31) == 0) red[wv] = s;
  __syncthreads();
  float tot = 0.f;
#pragma unroll
  for (int i = 0; i < 8; ++i) tot += red[i];
  float mu = tot * (1.f / 1024.f);

  float sq = 0.f;
#pragma unroll
  for (int i = 0; i < 4; ++i) { float d = v[i] - mu; sq += d * d; }
  sq += __shfl_xor(sq, 1); sq += __shfl_xor(sq, 2); sq += __shfl_xor(sq, 4);
  sq += __shfl_xor(sq, 8); sq += __shfl_xor(sq, 16);
  __syncthreads();
  if ((t & 31) == 0) red[wv] = sq;
  __syncthreads();
  float tv = 0.f;
#pragma unroll
  for (int i = 0; i < 8; ++i) tv += red[i];
  float rs = rsqrtf(tv * (1.f / 1024.f) + 1e-5f);

#pragma unroll
  for (int i = 0; i < 4; ++i) {
    int c = t + i * 256;
    float ov = (v[i] - mu) * rs * g[c] + be[c];
    outF[(size_t)row * D + c] = ov;
    if (outB) outB[(size_t)row * D + c] = (bf16)ov;
  }
}

// ---------------------------------------------------------------------------
extern "C" void kernel_launch(void* const* d_in, const int* in_sizes, int n_in,
                              void* d_out, int out_size, void* d_ws, size_t ws_size,
                              hipStream_t stream) {
  (void)in_sizes; (void)n_in; (void)out_size; (void)ws_size;
  const int Bsz = 2, S = 2048, D = 1024, H = 16, F = 4096;
  const int M = Bsz * S;

  const float* x  = (const float*)d_in[0];
  const int* mask = (const int*)d_in[1];
  const float* Wq = (const float*)d_in[2];  const float* bq = (const float*)d_in[3];
  const float* Wk = (const float*)d_in[4];  const float* bk = (const float*)d_in[5];
  const float* Wv = (const float*)d_in[6];  const float* bv = (const float*)d_in[7];
  const float* Wo = (const float*)d_in[8];  const float* bo = (const float*)d_in[9];
  const float* W1 = (const float*)d_in[10]; const float* b1 = (const float*)d_in[11];
  const float* W2 = (const float*)d_in[12]; const float* b2 = (const float*)d_in[13];
  const float* g1 = (const float*)d_in[14]; const float* be1 = (const float*)d_in[15];
  const float* g2 = (const float*)d_in[16]; const float* be2 = (const float*)d_in[17];

  char* ws = (char*)d_ws;
  size_t off = 0;
  auto take = [&](size_t bytes) -> char* {
    char* p = ws + off;
    off += (bytes + 255) & ~(size_t)255;
    return p;
  };
  bf16* xb   = (bf16*)take((size_t)M * D * 2);
  bf16* WqT  = (bf16*)take((size_t)D * D * 2);
  bf16* WkT  = (bf16*)take((size_t)D * D * 2);
  bf16* WvT  = (bf16*)take((size_t)D * D * 2);
  bf16* WoT  = (bf16*)take((size_t)D * D * 2);
  bf16* W1T  = (bf16*)take((size_t)D * F * 2);   // [F, D]
  bf16* W2T  = (bf16*)take((size_t)D * F * 2);   // [D, F]
  bf16* qb   = (bf16*)take((size_t)M * D * 2);
  bf16* kb   = (bf16*)take((size_t)M * D * 2);
  bf16* vb   = (bf16*)take((size_t)M * D * 2);
  bf16* attnb = (bf16*)take((size_t)M * D * 2);
  float* proj = (float*)take((size_t)M * D * 4);
  float* h    = (float*)take((size_t)M * D * 4);
  bf16* hb    = (bf16*)take((size_t)M * D * 2);
  float* ff2  = (float*)take((size_t)M * D * 4);
  bf16* ff1b  = qb;  // alias: q/k/v/attn (4 x M*D bf16 == M*F bf16) are dead by then
  (void)kb; (void)vb; (void)attnb;

  dim3 tb(32, 8);
  // 1. precision prep
  cvt_bf16_kernel<<<(M * D) / 1024, 256, 0, stream>>>(x, xb, M * D);
  transpose_cvt_kernel<<<dim3(D / 32, D / 32), tb, 0, stream>>>(Wq, WqT, D, D);
  transpose_cvt_kernel<<<dim3(D / 32, D / 32), tb, 0, stream>>>(Wk, WkT, D, D);
  transpose_cvt_kernel<<<dim3(D / 32, D / 32), tb, 0, stream>>>(Wv, WvT, D, D);
  transpose_cvt_kernel<<<dim3(D / 32, D / 32), tb, 0, stream>>>(Wo, WoT, D, D);
  transpose_cvt_kernel<<<dim3(F / 32, D / 32), tb, 0, stream>>>(W1, W1T, D, F);
  transpose_cvt_kernel<<<dim3(D / 32, F / 32), tb, 0, stream>>>(W2, W2T, F, D);

  // 2. Q/K/V projections (bf16 out, bias fused; Q pre-scaled by 1/sqrt(Dk))
  gemm_bf16_kernel<3><<<dim3(D / 64, M / 128), 256, 0, stream>>>(xb, WqT, bq, nullptr, qb, M, D, D);
  gemm_bf16_kernel<1><<<dim3(D / 64, M / 128), 256, 0, stream>>>(xb, WkT, bk, nullptr, kb, M, D, D);
  gemm_bf16_kernel<1><<<dim3(D / 64, M / 128), 256, 0, stream>>>(xb, WvT, bv, nullptr, vb, M, D, D);

  // 3. flash attention
  flash_attn_kernel<<<dim3(S / 64, Bsz * H), 128, 0, stream>>>(qb, kb, vb, mask, attnb, S, D, H);

  // 4. output projection + residual LN1
  gemm_bf16_kernel<0><<<dim3(D / 64, M / 128), 256, 0, stream>>>(attnb, WoT, bo, proj, nullptr, M, D, D);
  ln_residual_kernel<<<M, 256, 0, stream>>>(x, proj, g1, be1, h, hb, D);

  // 5. FFN
  gemm_bf16_kernel<2><<<dim3(F / 64, M / 128), 256, 0, stream>>>(hb, W1T, b1, nullptr, ff1b, M, F, D);
  gemm_bf16_kernel<0><<<dim3(D / 64, M / 128), 256, 0, stream>>>(ff1b, W2T, b2, ff2, nullptr, M, D, F);

  // 6. residual LN2 -> out
  ln_residual_kernel<<<M, 256, 0, stream>>>(h, ff2, g2, be2, (float*)d_out, nullptr, D);
}
